// CGConvNet_80324478369805
// MI455X (gfx1250) — matrix operand order repaired and checked by reading, not compiled, not run on hardware
//
#include <hip/hip_runtime.h>
#include <hip/hip_bf16.h>

#define N_NODES    100000
#define N_EDGES    1600000
#define IN_FEAT    64
#define EDGE_FEAT  16
#define NUM_CLASS  10
#define NUM_GRAPHS 64
#define ZDIM       144          // 2*IN_FEAT + EDGE_FEAT
#define KPAD       160          // K padded to 5 x 32 for bf16 WMMA
#define NTILES     (N_EDGES / 16)

typedef __attribute__((ext_vector_type(16))) __bf16        v16bf;
typedef __attribute__((ext_vector_type(8)))  float         v8f;
typedef __attribute__((ext_vector_type(8)))  unsigned int  v8u;

// fp32 -> bf16 round-to-nearest-even
__device__ __forceinline__ unsigned short f2bf(float f) {
    unsigned int u = __builtin_bit_cast(unsigned int, f);
    u += 0x7FFFu + ((u >> 16) & 1u);
    return (unsigned short)(u >> 16);
}

// ---------------------------------------------------------------------------
// Edge kernel: per 16-edge tile, compute
//   msg = sigmoid(z@W_f + b_f) * softplus(z@W_s + b_s)   (z: 16x144, W: 144x64)
// with bf16 WMMA, then atomic scatter-add into agg[dst].
// One wave (32 lanes) per tile; 4 waves / block; weights staged once / block.
// ---------------------------------------------------------------------------
__global__ __launch_bounds__(128)
void cgconv_edge_wmma(const float* __restrict__ x,
                      const long long* __restrict__ edge_index,
                      const float* __restrict__ edge_attr,
                      const float* __restrict__ W_f, const float* __restrict__ b_f,
                      const float* __restrict__ W_s, const float* __restrict__ b_s,
                      float* __restrict__ agg)
{
    __shared__ unsigned short sW[2][64 * KPAD];   // W_f^T, W_s^T  (n-major, bf16)
    __shared__ unsigned short sZ[4][16 * KPAD];   // per-wave z tile (bf16)
    __shared__ int            sIdx[4][32];        // per-wave: [0..15]=dst, [16..31]=src

    // ---- stage transposed weights in LDS as bf16, K zero-padded to 160 ----
    for (int i = threadIdx.x; i < ZDIM * 64; i += blockDim.x) {
        int k = i >> 6, n = i & 63;               // W[k][n] row-major
        sW[0][n * KPAD + k] = f2bf(W_f[i]);
        sW[1][n * KPAD + k] = f2bf(W_s[i]);
    }
    for (int i = threadIdx.x; i < 64 * (KPAD - ZDIM); i += blockDim.x) {
        int n = i >> 4, k = ZDIM + (i & 15);
        sW[0][n * KPAD + k] = 0;
        sW[1][n * KPAD + k] = 0;
    }
    __syncthreads();

    const int lane = threadIdx.x & 31;
    const int wv   = threadIdx.x >> 5;
    const int m16  = lane & 15;
    const int hi   = lane >> 4;
    unsigned short* zt  = sZ[wv];
    int*            idx = sIdx[wv];

    const int waveId = blockIdx.x * (blockDim.x >> 5) + wv;
    const int nWaves = gridDim.x * (blockDim.x >> 5);

    for (int t = waveId; t < NTILES; t += nWaves) {
        // -------- stage z tile: [x_dst(64) | x_src(64) | e_attr(16) | pad] ----
        if (lane < 16) {
            int e = t * 16 + lane;
            idx[lane]      = (int)edge_index[(long long)N_EDGES + e];   // dst (row 1)
            idx[16 + lane] = (int)edge_index[e];                        // src (row 0)
        }
        __builtin_amdgcn_wave_barrier();
        #pragma unroll 4
        for (int m = 0; m < 16; ++m) {
            int d = idx[m], s = idx[16 + m];
            const float2 xd = *(const float2*)&x[d * IN_FEAT + lane * 2];
            zt[m * KPAD + lane * 2]          = f2bf(xd.x);
            zt[m * KPAD + lane * 2 + 1]      = f2bf(xd.y);
            const float2 xs = *(const float2*)&x[s * IN_FEAT + lane * 2];
            zt[m * KPAD + 64 + lane * 2]     = f2bf(xs.x);
            zt[m * KPAD + 64 + lane * 2 + 1] = f2bf(xs.y);
            zt[m * KPAD + 128 + lane] =
                (lane < EDGE_FEAT) ? f2bf(edge_attr[(t * 16 + m) * EDGE_FEAT + lane])
                                   : (unsigned short)0;
        }
        __builtin_amdgcn_wave_barrier();

        // -------- WMMA main loop: 5 K-chunks x 4 N-tiles x 2 gates ----------
        v8f accF[4] = {};
        v8f accS[4] = {};
        const int koff = hi * 16;                     // A-layout K half-select
        const unsigned short* zr = &zt[m16 * KPAD];   // A row for this lane

        #pragma unroll
        for (int kc = 0; kc < 5; ++kc) {
            const int kb = kc * 32;
            // A fragment (16x32 bf16): lanes 0-15 -> K kb..kb+15, hi-lanes kb+16..kb+31
            v8u au;
            #pragma unroll
            for (int j = 0; j < 8; ++j) {
                int k0 = kb + koff + 2 * j;
                au[j] = *(const unsigned int*)&zr[k0];
            }
            v16bf afrag = __builtin_bit_cast(v16bf, au);

            #pragma unroll
            for (int nt = 0; nt < 4; ++nt) {
                const int n = nt * 16 + m16;
                // B fragment (32x16 bf16): lanes 0-15 hold K kb..kb+15, hi-lanes kb+16..kb+31
                const unsigned short* wf = &sW[0][n * KPAD + kb + hi * 16];
                const unsigned short* ws = &sW[1][n * KPAD + kb + hi * 16];
                v8u bu, cu;
                #pragma unroll
                for (int j = 0; j < 8; ++j) {
                    bu[j] = *(const unsigned int*)&wf[2 * j];
                    cu[j] = *(const unsigned int*)&ws[2 * j];
                }
                v16bf bF = __builtin_bit_cast(v16bf, bu);
                v16bf bS = __builtin_bit_cast(v16bf, cu);
                accF[nt] = __builtin_amdgcn_wmma_f32_16x16x32_bf16(
                    false, afrag, false, bF, (short)0, accF[nt], false, false);
                accS[nt] = __builtin_amdgcn_wmma_f32_16x16x32_bf16(
                    false, afrag, false, bS, (short)0, accS[nt], false, false);
            }
        }

        // -------- epilogue: bias + sigmoid*softplus, scatter-add to agg ------
        #pragma unroll
        for (int nt = 0; nt < 4; ++nt) {
            const int n   = nt * 16 + m16;
            const float bfv = b_f[n];
            const float bsv = b_s[n];
            #pragma unroll
            for (int v = 0; v < 8; ++v) {
                const int m = v + hi * 8;             // C/D layout: M = vgpr + hi*8
                float g = accF[nt][v] + bfv;
                float u = accS[nt][v] + bsv;
                float sig = 1.0f / (1.0f + __expf(-g));
                float sp  = (u > 20.0f) ? u : __logf(1.0f + __expf(u));
                atomicAdd(&agg[idx[m] * IN_FEAT + n], sig * sp);
            }
        }
    }
}

// ---------------------------------------------------------------------------
__global__ void zero_ws(float* __restrict__ p, int n) {
    int i = blockIdx.x * blockDim.x + threadIdx.x;
    if (i < n) p[i] = 0.0f;
}

__global__ void node_update_pool(const float* __restrict__ x, const float* __restrict__ agg,
                                 const long long* __restrict__ batch,
                                 float* __restrict__ sums, float* __restrict__ counts)
{
    int i = blockIdx.x * blockDim.x + threadIdx.x;
    if (i >= N_NODES * IN_FEAT) return;
    int node = i >> 6, f = i & 63;
    float h = x[i] + agg[i];
    h = h > 0.0f ? h : 0.0f;                 // relu(x + agg)
    int g = (int)batch[node];
    atomicAdd(&sums[g * IN_FEAT + f], h);
    if (f == 0) atomicAdd(&counts[g], 1.0f);
}

__global__ void head_kernel(const float* __restrict__ sums, const float* __restrict__ counts,
                            const float* __restrict__ lin_w, const float* __restrict__ lin_b,
                            float* __restrict__ out)
{
    int i = blockIdx.x * blockDim.x + threadIdx.x;
    if (i >= NUM_GRAPHS * NUM_CLASS) return;
    int g = i / NUM_CLASS, c = i % NUM_CLASS;
    float cnt = counts[g];
    if (cnt < 1.0f) cnt = 1.0f;              // jnp.maximum(counts, 1.0)
    float acc = 0.0f;
    #pragma unroll
    for (int f = 0; f < IN_FEAT; ++f)
        acc += sums[g * IN_FEAT + f] * lin_w[f * NUM_CLASS + c];
    out[i] = acc / cnt + lin_b[c];           // (sums/cnt)@W == (sums@W)/cnt
}

// ---------------------------------------------------------------------------
extern "C" void kernel_launch(void* const* d_in, const int* in_sizes, int n_in,
                              void* d_out, int out_size, void* d_ws, size_t ws_size,
                              hipStream_t stream)
{
    (void)in_sizes; (void)n_in; (void)out_size; (void)ws_size;

    const float*     x   = (const float*)d_in[0];
    const long long* ei  = (const long long*)d_in[1];
    const float*     ea  = (const float*)d_in[2];
    const long long* bat = (const long long*)d_in[3];
    const float*     W_f = (const float*)d_in[4];
    const float*     b_f = (const float*)d_in[5];
    const float*     W_s = (const float*)d_in[6];
    const float*     b_s = (const float*)d_in[7];
    const float*     lw  = (const float*)d_in[8];
    const float*     lb  = (const float*)d_in[9];
    float* out = (float*)d_out;

    // workspace layout: agg [N_NODES*64] | sums [64*64] | counts [64]
    float* agg    = (float*)d_ws;
    float* sums   = agg + (size_t)N_NODES * IN_FEAT;
    float* counts = sums + NUM_GRAPHS * IN_FEAT;

    const int nz = N_NODES * IN_FEAT + NUM_GRAPHS * IN_FEAT + NUM_GRAPHS;
    zero_ws<<<(nz + 255) / 256, 256, 0, stream>>>(agg, nz);

    cgconv_edge_wmma<<<2048, 128, 0, stream>>>(x, ei, ea, W_f, b_f, W_s, b_s, agg);

    node_update_pool<<<(N_NODES * IN_FEAT + 255) / 256, 256, 0, stream>>>(
        x, agg, bat, sums, counts);

    head_kernel<<<(NUM_GRAPHS * NUM_CLASS + 255) / 256, 256, 0, stream>>>(
        sums, counts, lw, lb, out);
}
